// CrossAttentionDecoder_23003844837484
// MI455X (gfx1250) — matrix operand order repaired and checked by reading, not compiled, not run on hardware
//
#include <hip/hip_runtime.h>

typedef __bf16 bf16;
typedef __attribute__((ext_vector_type(16))) bf16  v16bf;
typedef __attribute__((ext_vector_type(8)))  bf16  v8bf;
typedef __attribute__((ext_vector_type(8)))  float v8f;

#define S_   16
#define B_   16
#define C_   256
#define H_   4
#define DH_  64
#define HW_  293
#define LQ_  (S_ * HW_)      // 4688
#define LK_  128
#define RQ_  (LQ_ * B_)      // 75008 (multiple of 128)
#define RK_  (LK_ * B_)      // 2048

// output region offsets (floats)
#define MODE_OFF  0
#define SHAPE_OFF 5120
#define COLOR_OFF 21504
#define TX_OFF    25600
#define HX_OFF    2122752

__device__ __forceinline__ bf16 f2bf(float f) {
  unsigned u = __builtin_bit_cast(unsigned, f);
  unsigned r = u + 0x7FFFu + ((u >> 16) & 1u);
  unsigned short s = (unsigned short)(r >> 16);
  return __builtin_bit_cast(bf16, s);
}
__device__ __forceinline__ float bf2f(bf16 b) {
  unsigned short s = __builtin_bit_cast(unsigned short, b);
  unsigned u = ((unsigned)s) << 16;
  return __builtin_bit_cast(float, u);
}

__device__ __forceinline__ v8f wmma_bf(v16bf a, v16bf b, v8f c) {
  return __builtin_amdgcn_wmma_f32_16x16x32_bf16(false, a, false, b, (short)0, c, false, false);
}

// A fragment (16x32 bf16): lane holds row M = m0 + lane%16.
// lanes 0-15: K 0..7 and 16..23 ; lanes 16-31: K 8..15 and 24..31
__device__ __forceinline__ v16bf ld_a(const bf16* A, int lda, int m0, int k0, int mclamp) {
  int lane = threadIdx.x & 31;
  int row = m0 + (lane & 15);
  if (row > mclamp) row = mclamp;
  int hi = (lane >> 4) & 1;
  const bf16* p = A + (size_t)row * lda + k0 + hi * 8;
  v8bf a0 = *(const v8bf*)p;
  v8bf a1 = *(const v8bf*)(p + 16);
  v16bf r;
#pragma unroll
  for (int i = 0; i < 8; i++) { r[i] = a0[i]; r[i + 8] = a1[i]; }
  return r;
}

// B fragment (32x16 bf16) from transposed weights WT[N][K]:
// lane holds col N = n0 + lane%16; lanes 0-15: K 0..7 and 16..23 ; lanes 16-31: K 8..15 and 24..31
__device__ __forceinline__ v16bf ld_b(const bf16* WT, int ldk, int n0, int k0) {
  int lane = threadIdx.x & 31;
  int n = n0 + (lane & 15);
  int kk = k0 + ((lane >> 4) & 1) * 8;
  const bf16* p = WT + (size_t)n * ldk + kk;
  v8bf b0 = *(const v8bf*)p;
  v8bf b1 = *(const v8bf*)(p + 16);
  v16bf r;
#pragma unroll
  for (int i = 0; i < 8; i++) { r[i] = b0[i]; r[i + 8] = b1[i]; }
  return r;
}

// ---------------- weight prep ----------------
__global__ void xpose_bf(bf16* __restrict__ d, const float* __restrict__ s, int K, int N) {
  int tot = K * N;
  for (int i = blockIdx.x * blockDim.x + threadIdx.x; i < tot; i += gridDim.x * blockDim.x) {
    int n = i / K, k = i % K;
    d[i] = f2bf(s[(size_t)k * N + n]);   // d[n][k] = s[k][n]
  }
}
__global__ void conv_bf(bf16* __restrict__ d, const float* __restrict__ s, int n) {
  for (int i = blockIdx.x * blockDim.x + threadIdx.x; i < n; i += gridDim.x * blockDim.x)
    d[i] = f2bf(s[i]);
}

// ---------------- LayerNorm kernels ----------------
__device__ __forceinline__ void blk_ln(float v, float g, float b, bf16* dst) {
  __shared__ float r1[8], r2[8];
  float sv = v, sq = v * v;
#pragma unroll
  for (int off = 16; off; off >>= 1) { sv += __shfl_xor(sv, off, 32); sq += __shfl_xor(sq, off, 32); }
  if ((threadIdx.x & 31) == 0) { r1[threadIdx.x >> 5] = sv; r2[threadIdx.x >> 5] = sq; }
  __syncthreads();
  float tot = 0.f, tot2 = 0.f;
#pragma unroll
  for (int i = 0; i < 8; i++) { tot += r1[i]; tot2 += r2[i]; }
  float mean = tot * (1.f / 256.f);
  float var = tot2 * (1.f / 256.f) - mean * mean;
  float rstd = rsqrtf(var + 1e-5f);
  *dst = f2bf((v - mean) * rstd * g + b);
}

__global__ void ln_qn(const int* __restrict__ tq, const float* __restrict__ temporal,
                      const float* __restrict__ spatial, const float* __restrict__ g,
                      const float* __restrict__ bb, bf16* __restrict__ qn) {
  int r = blockIdx.x; int l = r >> 4; int b = r & 15;
  int s = l / HW_; int hw = l % HW_;
  int t = tq[s * B_ + b];
  int c = threadIdx.x;
  float v = temporal[(size_t)t * C_ + c] + spatial[(size_t)hw * C_ + c];
  blk_ln(v, g[c], bb[c], &qn[(size_t)r * C_ + c]);
}

__global__ void ln2k(const float* __restrict__ x, const float* __restrict__ g,
                     const float* __restrict__ bb, bf16* __restrict__ o) {
  int r = blockIdx.x; int c = threadIdx.x;
  float v = x[(size_t)r * C_ + c];
  blk_ln(v, g[c], bb[c], &o[(size_t)r * C_ + c]);
}

// ---------------- generic WMMA GEMM: out = A(MxK) * W(KxN) + bias ----------------
// WT is W transposed [N][K] bf16.  mode 0: store bf16. mode 1: store f32 + xq residual.
__global__ void gemm_bf(const bf16* __restrict__ A, const bf16* __restrict__ WT,
                        const float* __restrict__ bias, bf16* __restrict__ outB,
                        float* __restrict__ outF, int M, int N, int K, int mode,
                        const int* __restrict__ tq, const float* __restrict__ temporal,
                        const float* __restrict__ spatial) {
  int w = threadIdx.x >> 5, lane = threadIdx.x & 31;
  int ln = lane & 15, hi = lane >> 4;
  int m0 = blockIdx.x * 128 + (w & 3) * 32;
  int n0 = blockIdx.y * 64 + (w >> 2) * 32;
  v8f acc[2][2] = {};
  for (int k0 = 0; k0 < K; k0 += 32) {
    v16bf a0 = ld_a(A, K, m0, k0, M - 1);
    v16bf a1 = ld_a(A, K, m0 + 16, k0, M - 1);
    v16bf b0 = ld_b(WT, K, n0, k0);
    v16bf b1 = ld_b(WT, K, n0 + 16, k0);
    acc[0][0] = wmma_bf(a0, b0, acc[0][0]);
    acc[0][1] = wmma_bf(a0, b1, acc[0][1]);
    acc[1][0] = wmma_bf(a1, b0, acc[1][0]);
    acc[1][1] = wmma_bf(a1, b1, acc[1][1]);
  }
#pragma unroll
  for (int ti = 0; ti < 2; ti++)
#pragma unroll
    for (int tj = 0; tj < 2; tj++)
#pragma unroll
      for (int e = 0; e < 8; e++) {
        int r = m0 + ti * 16 + e + 8 * hi;
        int n = n0 + tj * 16 + ln;
        float v = acc[ti][tj][e] + bias[n];
        if (mode == 0) {
          outB[(size_t)r * N + n] = f2bf(v);
        } else {
          int l = r >> 4, b = r & 15;
          int s = l / HW_, hw = l % HW_;
          int t = tq[s * B_ + b];
          v += temporal[(size_t)t * C_ + n] + spatial[(size_t)hw * C_ + n];
          outF[(size_t)r * N + n] = v;
        }
      }
}

// ---------------- attention: one (b,h) and 64 query rows per block ----------------
__global__ void attn_kernel(const bf16* __restrict__ Qb, const bf16* __restrict__ Kb,
                            const bf16* __restrict__ Vb, const int* __restrict__ tq,
                            const int* __restrict__ tk, const int* __restrict__ pad_q,
                            const int* __restrict__ pad_k, bf16* __restrict__ Ab) {
  __shared__ __align__(16) bf16 Ksh[LK_ * DH_];    // 16KB, reused as P (64x128 bf16)
  __shared__ __align__(16) bf16 VshT[DH_ * LK_];   // 16KB, V transposed [d][k]
  __shared__ float smax[2][64];
  __shared__ float ssum[2][64];

  int bh = blockIdx.y; int b = bh >> 2; int h = bh & 3;
  int qbase = blockIdx.x * 64;
  int w = threadIdx.x >> 5, lane = threadIdx.x & 31;
  int ln = lane & 15, hi = lane >> 4;

  for (int i = threadIdx.x; i < LK_ * DH_; i += 256) {
    int k = i >> 6, d = i & 63;
    size_t src = ((size_t)k * B_ + b) * C_ + h * DH_ + d;
    Ksh[i] = Kb[src];
    VshT[d * LK_ + k] = Vb[src];
  }
  __syncthreads();

  // scores: wave w -> q rows qr0..qr0+15, key cols kc0..kc0+63
  int qr0 = (w & 3) * 16;
  int kc0 = (w >> 2) * 64;
  const bf16* Qbase = Qb + (size_t)b * C_ + h * DH_;
  v8f acc[4] = {};
  for (int k0 = 0; k0 < DH_; k0 += 32) {
    v16bf a = ld_a(Qbase, B_ * C_, qbase + qr0, k0, LQ_ - 1);
#pragma unroll
    for (int t = 0; t < 4; t++) {
      v16bf bf = ld_b(Ksh, DH_, kc0 + 16 * t, k0);
      acc[t] = wmma_bf(a, bf, acc[t]);
    }
  }

  // scale + mask in place
  int padq = pad_q[b] * HW_;
  int pkb = pad_k[b];
#pragma unroll
  for (int t = 0; t < 4; t++) {
    int kcol = kc0 + 16 * t + ln;
    int tkv = tk[kcol * B_ + b];
    bool kok = (kcol < pkb);
#pragma unroll
    for (int e = 0; e < 8; e++) {
      int l = qbase + qr0 + e + 8 * hi;
      float v = acc[t][e] * 0.125f;
      bool ok = kok;
      if (l < LQ_) {
        int s = l / HW_;
        ok = ok && (tkv <= tq[s * B_ + b]) && (l < padq);
      }
      if (!ok) v = -1e9f;
      acc[t][e] = v;
    }
  }

  // softmax over 128 keys: 16-lane shuffle reduce + cross-half via LDS
  int half = w >> 2;
#pragma unroll
  for (int e = 0; e < 8; e++) {
    float m = fmaxf(fmaxf(acc[0][e], acc[1][e]), fmaxf(acc[2][e], acc[3][e]));
#pragma unroll
    for (int off = 1; off < 16; off <<= 1) m = fmaxf(m, __shfl_xor(m, off, 32));
    if (ln == 0) smax[half][qr0 + e + 8 * hi] = m;
  }
  __syncthreads();
#pragma unroll
  for (int e = 0; e < 8; e++) {
    int row = qr0 + e + 8 * hi;
    float m = fmaxf(smax[0][row], smax[1][row]);
    float s = 0.f;
#pragma unroll
    for (int t = 0; t < 4; t++) { float ev = __expf(acc[t][e] - m); acc[t][e] = ev; s += ev; }
#pragma unroll
    for (int off = 1; off < 16; off <<= 1) s += __shfl_xor(s, off, 32);
    if (ln == 0) ssum[half][row] = s;
  }
  __syncthreads();
  bf16* Pbf = Ksh;  // reuse (K tile dead)
#pragma unroll
  for (int e = 0; e < 8; e++) {
    int row = qr0 + e + 8 * hi;
    float rinv = 1.f / (ssum[0][row] + ssum[1][row]);
#pragma unroll
    for (int t = 0; t < 4; t++) {
      int kcol = kc0 + 16 * t + ln;
      Pbf[row * LK_ + kcol] = f2bf(acc[t][e] * rinv);
    }
  }
  __syncthreads();

  // out = P(64x128) @ V(128x64): wave w -> rows qr0, d cols dc0..dc0+31
  int dc0 = (w >> 2) * 32;
  v8f oacc[2] = {};
  for (int k0 = 0; k0 < LK_; k0 += 32) {
    v16bf a = ld_a(Pbf, LK_, qr0, k0, 63);
#pragma unroll
    for (int t = 0; t < 2; t++) {
      v16bf bf = ld_b(VshT, LK_, dc0 + 16 * t, k0);
      oacc[t] = wmma_bf(a, bf, oacc[t]);
    }
  }
#pragma unroll
  for (int t = 0; t < 2; t++)
#pragma unroll
    for (int e = 0; e < 8; e++) {
      int l = qbase + qr0 + e + 8 * hi;
      if (l < LQ_) {
        int d = dc0 + 16 * t + ln;
        Ab[((size_t)l * B_ + b) * C_ + h * DH_ + d] = f2bf(oacc[t][e]);
      }
    }
}

// ---------------- fused MLP: x2 = x + gelu(ln2(x)@w1+b1)@w2+b2, per 64-row tile ----------------
__global__ void mlp_fused(const bf16* __restrict__ xln, const float* __restrict__ xres,
                          const bf16* __restrict__ w1T, const float* __restrict__ b1,
                          const bf16* __restrict__ w2T, const float* __restrict__ b2,
                          bf16* __restrict__ x2) {
  __shared__ __align__(16) bf16 Hsh[64 * 256];   // 32KB hidden tile
  int w = threadIdx.x >> 5, lane = threadIdx.x & 31;
  int ln = lane & 15, hi = lane >> 4;
  int m0 = blockIdx.x * 64;
  int r0 = (w & 3) * 16;     // 16-row slab in tile
  int c0 = (w >> 2) * 128;   // 128-col slab
  v8f oacc[8] = {};
  for (int j = 0; j < 4; j++) {   // hidden chunks of 256
    v8f hacc[8] = {};
    for (int k0 = 0; k0 < 256; k0 += 32) {
      v16bf a = ld_a(xln, 256, m0 + r0, k0, RQ_ - 1);
#pragma unroll
      for (int t = 0; t < 8; t++) {
        v16bf bf = ld_b(w1T, 256, j * 256 + c0 + 16 * t, k0);
        hacc[t] = wmma_bf(a, bf, hacc[t]);
      }
    }
#pragma unroll
    for (int t = 0; t < 8; t++)
#pragma unroll
      for (int e = 0; e < 8; e++) {
        int cc = c0 + 16 * t + ln;
        int rr = r0 + e + 8 * hi;
        float xv = hacc[t][e] + b1[j * 256 + cc];
        float gl = 0.5f * xv * (1.f + tanhf(0.7978845608f * (xv + 0.044715f * xv * xv * xv)));
        Hsh[rr * 256 + cc] = f2bf(gl);
      }
    __syncthreads();
    for (int k0 = 0; k0 < 256; k0 += 32) {
      v16bf a = ld_a(Hsh, 256, r0, k0, 63);
#pragma unroll
      for (int t = 0; t < 8; t++) {
        v16bf bf = ld_b(w2T, 1024, c0 + 16 * t, j * 256 + k0);
        oacc[t] = wmma_bf(a, bf, oacc[t]);
      }
    }
    __syncthreads();
  }
#pragma unroll
  for (int t = 0; t < 8; t++)
#pragma unroll
    for (int e = 0; e < 8; e++) {
      int rr = m0 + r0 + e + 8 * hi;
      int cc = c0 + 16 * t + ln;
      float v = oacc[t][e] + b2[cc] + xres[(size_t)rr * 256 + cc];
      x2[(size_t)rr * 256 + cc] = f2bf(v);
    }
}

// ---------------- table / hand heads with pixel-shuffle scatter ----------------
__global__ void head_gemm(const bf16* __restrict__ X, const bf16* __restrict__ WT,
                          const float* __restrict__ bias, float* __restrict__ out, int which) {
  int w = threadIdx.x >> 5, lane = threadIdx.x & 31;
  int ln = lane & 15, hi = lane >> 4;
  int m0 = blockIdx.x * 128 + w * 16;          // 16 rows == one (s,hw), b = 0..15
  int nhw = which ? 36 : 256;
  int idx16 = m0 >> 4;
  int hw = idx16 % nhw;
  int s = idx16 / nhw;
  int hwAbs = which ? (256 + hw) : hw;
  const bf16* base = X + ((size_t)(s * HW_ + hwAbs) * B_) * C_;
  v8f acc[2] = {};
  for (int k0 = 0; k0 < 256; k0 += 32) {
    v16bf a = ld_a(base, C_, 0, k0, 15);
#pragma unroll
    for (int t = 0; t < 2; t++) {
      v16bf bf = ld_b(WT, C_, 16 * t, k0);
      acc[t] = wmma_bf(a, bf, acc[t]);
    }
  }
#pragma unroll
  for (int t = 0; t < 2; t++)
#pragma unroll
    for (int e = 0; e < 8; e++) {
      int b = e + 8 * hi;
      int n = 16 * t + ln;
      float v = acc[t][e] + bias[n];
      int uh = n >> 3, uw = (n >> 1) & 3, uc = n & 1;
      size_t off;
      if (which == 0) {
        int th = hw >> 4, tw = hw & 15;
        off = TX_OFF + ((((size_t)(s * 16 + b) * 2 + uc) * 64 + th * 4 + uh) * 64 + tw * 4 + uw);
      } else {
        int hh = hw / 6, ww = hw % 6;
        off = HX_OFF + ((((size_t)(s * 16 + b) * 2 + uc) * 24 + hh * 4 + uh) * 24 + ww * 4 + uw);
      }
      out[off] = v;
    }
}

// ---------------- global token heads (mode/shape/color) ----------------
__global__ void gx_head(const bf16* __restrict__ X, const float* __restrict__ mw,
                        const float* __restrict__ mb, const float* __restrict__ sw,
                        const float* __restrict__ sb, const float* __restrict__ cw,
                        const float* __restrict__ cb, float* __restrict__ out) {
  int m = blockIdx.x; int s = m >> 4; int b = m & 15;
  size_t row = ((size_t)(s * HW_ + HW_ - 1) * B_ + b) * C_;
  __shared__ float xr[256];
  for (int i = threadIdx.x; i < 256; i += blockDim.x) xr[i] = bf2f(X[row + i]);
  __syncthreads();
  for (int o = threadIdx.x; o < 100; o += blockDim.x) {
    const float* W; const float* Bb; int N; int n; size_t ob;
    if (o < 20)      { W = mw; Bb = mb; N = 20; n = o;      ob = MODE_OFF; }
    else if (o < 84) { W = sw; Bb = sb; N = 64; n = o - 20; ob = SHAPE_OFF; }
    else             { W = cw; Bb = cb; N = 16; n = o - 84; ob = COLOR_OFF; }
    float acc = Bb[n];
    for (int c = 0; c < 256; c++) acc += xr[c] * W[(size_t)c * N + n];
    out[ob + (size_t)m * N + n] = acc;
  }
}

extern "C" void kernel_launch(void* const* d_in, const int* in_sizes, int n_in,
                              void* d_out, int out_size, void* d_ws, size_t ws_size,
                              hipStream_t stream) {
  (void)in_sizes; (void)n_in; (void)out_size; (void)ws_size;
  const int*   tq   = (const int*)  d_in[0];
  const int*   padq = (const int*)  d_in[1];
  const float* xk   = (const float*)d_in[2];
  const int*   tk   = (const int*)  d_in[3];
  const int*   padk = (const int*)  d_in[4];
  const float* spat = (const float*)d_in[8];
  const float* temp = (const float*)d_in[9];
  const float* ln1g = (const float*)d_in[10];
  const float* ln1b = (const float*)d_in[11];
  const float* wq = (const float*)d_in[12]; const float* bq = (const float*)d_in[13];
  const float* wk = (const float*)d_in[14]; const float* bk = (const float*)d_in[15];
  const float* wv = (const float*)d_in[16]; const float* bv = (const float*)d_in[17];
  const float* wo = (const float*)d_in[18]; const float* bo = (const float*)d_in[19];
  const float* ln2g = (const float*)d_in[20]; const float* ln2b = (const float*)d_in[21];
  const float* w1 = (const float*)d_in[22]; const float* b1 = (const float*)d_in[23];
  const float* w2 = (const float*)d_in[24]; const float* b2 = (const float*)d_in[25];
  const float* tw = (const float*)d_in[26]; const float* tb = (const float*)d_in[27];
  const float* hwt = (const float*)d_in[28]; const float* hb = (const float*)d_in[29];
  const float* mw = (const float*)d_in[30]; const float* mb = (const float*)d_in[31];
  const float* sw = (const float*)d_in[32]; const float* sb = (const float*)d_in[33];
  const float* cw = (const float*)d_in[34]; const float* cb = (const float*)d_in[35];
  float* out = (float*)d_out;

  char* ws = (char*)d_ws;
  size_t off = 0;
  auto alloc = [&](size_t bytes) -> char* {
    char* p = ws + off;
    off = (off + bytes + 255) & ~(size_t)255;
    return p;
  };
  bf16* wqT  = (bf16*)alloc(256 * 256 * 2);
  bf16* wkT  = (bf16*)alloc(256 * 256 * 2);
  bf16* wvT  = (bf16*)alloc(256 * 256 * 2);
  bf16* woT  = (bf16*)alloc(256 * 256 * 2);
  bf16* w1T  = (bf16*)alloc(1024 * 256 * 2);
  bf16* w2T  = (bf16*)alloc(256 * 1024 * 2);
  bf16* tabT = (bf16*)alloc(32 * 256 * 2);
  bf16* hanT = (bf16*)alloc(32 * 256 * 2);
  bf16* xkb  = (bf16*)alloc((size_t)RK_ * C_ * 2);
  bf16* qnb  = (bf16*)alloc((size_t)RQ_ * C_ * 2);
  bf16* Qb   = (bf16*)alloc((size_t)RQ_ * C_ * 2);
  bf16* Kb   = (bf16*)alloc((size_t)RK_ * C_ * 2);
  bf16* Vb   = (bf16*)alloc((size_t)RK_ * C_ * 2);
  bf16* ab   = (bf16*)alloc((size_t)RQ_ * C_ * 2);
  float* xf  = (float*)alloc((size_t)RQ_ * C_ * 4);
  bf16* xlnb = (bf16*)alloc((size_t)RQ_ * C_ * 2);
  bf16* x2b  = (bf16*)alloc((size_t)RQ_ * C_ * 2);

  // 1) weight prep
  xpose_bf<<<dim3(256), dim3(256), 0, stream>>>(wqT, wq, 256, 256);
  xpose_bf<<<dim3(256), dim3(256), 0, stream>>>(wkT, wk, 256, 256);
  xpose_bf<<<dim3(256), dim3(256), 0, stream>>>(wvT, wv, 256, 256);
  xpose_bf<<<dim3(256), dim3(256), 0, stream>>>(woT, wo, 256, 256);
  xpose_bf<<<dim3(512), dim3(256), 0, stream>>>(w1T, w1, 256, 1024);
  xpose_bf<<<dim3(512), dim3(256), 0, stream>>>(w2T, w2, 1024, 256);
  xpose_bf<<<dim3(32),  dim3(256), 0, stream>>>(tabT, tw, 256, 32);
  xpose_bf<<<dim3(32),  dim3(256), 0, stream>>>(hanT, hwt, 256, 32);
  conv_bf<<<dim3(512), dim3(256), 0, stream>>>(xkb, xk, RK_ * C_);

  // 2) qn = LN1(xq) as bf16
  ln_qn<<<dim3(RQ_), dim3(256), 0, stream>>>(tq, temp, spat, ln1g, ln1b, qnb);

  // 3) Q/K/V projections (WMMA GEMMs)
  gemm_bf<<<dim3(RQ_ / 128, 4), dim3(256), 0, stream>>>(qnb, wqT, bq, Qb, nullptr,
      RQ_, 256, 256, 0, nullptr, nullptr, nullptr);
  gemm_bf<<<dim3(RK_ / 128, 4), dim3(256), 0, stream>>>(xkb, wkT, bk, Kb, nullptr,
      RK_, 256, 256, 0, nullptr, nullptr, nullptr);
  gemm_bf<<<dim3(RK_ / 128, 4), dim3(256), 0, stream>>>(xkb, wvT, bv, Vb, nullptr,
      RK_, 256, 256, 0, nullptr, nullptr, nullptr);

  // 4) masked attention
  attn_kernel<<<dim3((LQ_ + 63) / 64, B_ * H_), dim3(256), 0, stream>>>(
      Qb, Kb, Vb, tq, tk, padq, padk, ab);

  // 5) output projection + xq residual -> x (f32)
  gemm_bf<<<dim3(RQ_ / 128, 4), dim3(256), 0, stream>>>(ab, woT, bo, nullptr, xf,
      RQ_, 256, 256, 1, tq, temp, spat);

  // 6) LN2 -> bf16
  ln2k<<<dim3(RQ_), dim3(256), 0, stream>>>(xf, ln2g, ln2b, xlnb);

  // 7) fused MLP + residual -> x2 (bf16)
  mlp_fused<<<dim3(RQ_ / 64), dim3(256), 0, stream>>>(xlnb, xf, w1T, b1, w2T, b2, x2b);

  // 8) heads
  head_gemm<<<dim3((S_ * 256 * B_) / 128), dim3(256), 0, stream>>>(x2b, tabT, tb, out, 0);
  head_gemm<<<dim3((S_ * 36 * B_) / 128),  dim3(256), 0, stream>>>(x2b, hanT, hb, out, 1);
  gx_head<<<dim3(S_ * B_), dim3(128), 0, stream>>>(x2b, mw, mb, sw, sb, cw, cb, out);
}
